// DeformerAE_32014686224762
// MI455X (gfx1250) — matrix operand order repaired and checked by reading, not compiled, not run on hardware
//
#include <hip/hip_runtime.h>
#include <hip/hip_bf16.h>
#include <stdint.h>

// ---------------------------------------------------------------------------
// PointNet++ SSG backbone for MI455X (gfx1250, wave32, WMMA).
// B=32, N=2048.  All GEMMs run on v_wmma_f32_16x16x32_f16 (f16 in, f32 acc).
// BN (batch statistics) is folded as a packed-f16 per-channel affine into the
// consumer GEMM's A-load / the max-pool; raw activations written once in f16.
// Weights are pre-converted to zero-padded f16 so the inner loop is pure
// 128-bit vector loads + v_wmma (no bounds checks, EXEC all-ones).
// ---------------------------------------------------------------------------

typedef _Float16 v16h __attribute__((ext_vector_type(16)));
typedef _Float16 v8h  __attribute__((ext_vector_type(8)));
typedef float    v8f  __attribute__((ext_vector_type(8)));

#define BB 32
#define NN 2048

// ------------------------------ utility ------------------------------------

__global__ void zero_f32_kernel(float* __restrict__ p, int n) {
  int i = blockIdx.x * blockDim.x + threadIdx.x;
  if (i < n) p[i] = 0.0f;
}

// xyz (B,3,N) -> pts (B,N,3)
__global__ void transpose_xyz_kernel(const float* __restrict__ xyz,
                                     float* __restrict__ pts) {
  int i = blockIdx.x * blockDim.x + threadIdx.x;  // over B*3*N
  if (i >= BB * 3 * NN) return;
  int b = i / (3 * NN);
  int r = i - b * 3 * NN;
  int c = r / NN;
  int j = r - c * NN;
  pts[(b * NN + j) * 3 + c] = xyz[i];
}

// W (Cout x Cin, f32) -> Wp (Cout x Kp, f16, zero padded)
__global__ void wprep_kernel(const float* __restrict__ W, int Cin, int Kp,
                             int Cout, _Float16* __restrict__ Wp) {
  int i = blockIdx.x * blockDim.x + threadIdx.x;
  if (i >= Cout * Kp) return;
  int o = i / Kp;
  int k = i - o * Kp;
  Wp[i] = (_Float16)((k < Cin) ? W[(size_t)o * Cin + k] : 0.0f);
}

// ------------------------- farthest point sample ---------------------------
__global__ __launch_bounds__(256)
void fps_kernel(const float* __restrict__ pts, int n, int npoint,
                int* __restrict__ out_idx, float* __restrict__ new_xyz) {
  __shared__ float dist[2048];
  __shared__ float rv[256];
  __shared__ int   ri[256];
  __shared__ int   s_sel;

  const int b = blockIdx.x;
  const int tid = threadIdx.x;
  const float* p = pts + (size_t)b * n * 3;

  for (int j = tid; j < n; j += 256) dist[j] = 1e10f;
  if (tid == 0) s_sel = 0;
  __syncthreads();

  for (int i = 0; i < npoint; ++i) {
    int sel = s_sel;
    if (tid == 0) {
      out_idx[b * npoint + i] = sel;
      new_xyz[(size_t)(b * npoint + i) * 3 + 0] = p[sel * 3 + 0];
      new_xyz[(size_t)(b * npoint + i) * 3 + 1] = p[sel * 3 + 1];
      new_xyz[(size_t)(b * npoint + i) * 3 + 2] = p[sel * 3 + 2];
    }
    float cx = p[sel * 3 + 0], cy = p[sel * 3 + 1], cz = p[sel * 3 + 2];
    float best = -1.0f; int bi = 0;
    for (int j = tid; j < n; j += 256) {
      float dx = p[j * 3 + 0] - cx;
      float dy = p[j * 3 + 1] - cy;
      float dz = p[j * 3 + 2] - cz;
      float d = dx * dx + dy * dy + dz * dz;
      float nd = fminf(dist[j], d);
      dist[j] = nd;
      if (nd > best) { best = nd; bi = j; }   // j ascending -> first max kept
    }
    rv[tid] = best; ri[tid] = bi;
    __syncthreads();
    for (int s2 = 128; s2 > 0; s2 >>= 1) {
      if (tid < s2) {
        float ov = rv[tid + s2]; int oi = ri[tid + s2];
        if (ov > rv[tid] || (ov == rv[tid] && oi < ri[tid])) { rv[tid] = ov; ri[tid] = oi; }
      }
      __syncthreads();
    }
    if (tid == 0) s_sel = ri[0];
    __syncthreads();
  }
}

// ------------------------------ ball query ---------------------------------
__global__ __launch_bounds__(32)
void ball_query_kernel(const float* __restrict__ pts,
                       const float* __restrict__ new_xyz,
                       int n, int S, int nsample, float r2,
                       int* __restrict__ out) {
  const int q = blockIdx.x;          // over B*S
  const int b = q / S;
  const int lane = threadIdx.x;
  const float* p = pts + (size_t)b * n * 3;
  const float cx = new_xyz[(size_t)q * 3 + 0];
  const float cy = new_xyz[(size_t)q * 3 + 1];
  const float cz = new_xyz[(size_t)q * 3 + 2];
  int* o = out + (size_t)q * nsample;

  int count = 0;
  int first = -1;
  for (int base = 0; base < n && count < nsample; base += 32) {
    int j = base + lane;
    bool ok = false;
    if (j < n) {
      float dx = p[j * 3 + 0] - cx;
      float dy = p[j * 3 + 1] - cy;
      float dz = p[j * 3 + 2] - cz;
      ok = (dx * dx + dy * dy + dz * dz) <= r2;
    }
    unsigned mask = __builtin_amdgcn_ballot_w32(ok);
    int pre = __popc(mask & ((1u << lane) - 1u));
    int pos = count + pre;
    if (ok && pos < nsample) o[pos] = j;
    if (first < 0 && mask != 0u) first = base + __builtin_ctz(mask);
    count += __popc(mask);
  }
  if (first < 0) first = 0;
  if (count < nsample) {
    for (int t = count + lane; t < nsample; t += 32) o[t] = first;
  }
}

// ----------------------------- grouping ------------------------------------
__global__ void group_kernel(const float* __restrict__ pts,
                             const float* __restrict__ feats, int featC,
                             const float* __restrict__ new_xyz,
                             const int* __restrict__ idx,
                             int S, int Kn, int Kp, int n, long long R,
                             _Float16* __restrict__ G) {
  for (long long r = (long long)blockIdx.x * blockDim.x + threadIdx.x; r < R;
       r += (long long)gridDim.x * blockDim.x) {
    int q = (int)(r / Kn);
    int b = q / S;
    int j = idx[r];
    const float* pj = pts + ((size_t)b * n + j) * 3;
    _Float16* g = G + (size_t)r * Kp;
    g[0] = (_Float16)(pj[0] - new_xyz[(size_t)q * 3 + 0]);
    g[1] = (_Float16)(pj[1] - new_xyz[(size_t)q * 3 + 1]);
    g[2] = (_Float16)(pj[2] - new_xyz[(size_t)q * 3 + 2]);
    const float* f = feats + ((size_t)b * n + j) * featC;
    for (int c = 0; c < featC; ++c) g[3 + c] = (_Float16)f[c];
    for (int c = 3 + featC; c < Kp; ++c) g[c] = (_Float16)0.0f;
  }
}

__global__ void group_all_kernel(const float* __restrict__ gxyz,
                                 const float* __restrict__ gfeat, int featC,
                                 int Kp, long long R, _Float16* __restrict__ G) {
  for (long long r = (long long)blockIdx.x * blockDim.x + threadIdx.x; r < R;
       r += (long long)gridDim.x * blockDim.x) {
    _Float16* g = G + (size_t)r * Kp;
    g[0] = (_Float16)gxyz[(size_t)r * 3 + 0];
    g[1] = (_Float16)gxyz[(size_t)r * 3 + 1];
    g[2] = (_Float16)gxyz[(size_t)r * 3 + 2];
    const float* f = gfeat + (size_t)r * featC;
    for (int c = 0; c < featC; ++c) g[3 + c] = (_Float16)f[c];
    for (int c = 3 + featC; c < Kp; ++c) g[c] = (_Float16)0.0f;
  }
}

// ------------------------------- GEMM --------------------------------------
// Y[r, n] = bias[n] + sum_k act(A[r,k]) * Wp[n,k]
// FUSED: act = ReLU(a*scale[k]+shift[k]) in packed f16 (BN of previous layer).
// One wave per 16x16 tile; K stepped by 32 through v_wmma_f32_16x16x32_f16.
// A-fragment (ISA 7.12.2, 16-bit A 16x32): lane's 16 elements are exactly the
// two contiguous 8-half chunks at row*lda + k0 + 8*half and +16.
// B-fragment: two contiguous 8-half chunks at ncol*lda + k0 + 8*half and +16.
// R is a multiple of 16 and lda a multiple of 32 -> no bounds checks at all.
template <bool FUSED>
__global__ __launch_bounds__(128)
void gemm_wmma_kernel(const _Float16* __restrict__ A, int lda,
                      const _Float16* __restrict__ sch,
                      const _Float16* __restrict__ shh,
                      const _Float16* __restrict__ Wp,
                      const float* __restrict__ bias,
                      int R, int Cout, _Float16* __restrict__ Y) {
  const int wave = threadIdx.x >> 5;
  const int lane = threadIdx.x & 31;
  const int mt = blockIdx.x * (blockDim.x >> 5) + wave;
  const int m0 = mt * 16;
  if (m0 >= R) return;                       // wave-uniform
  const int n0 = blockIdx.y * 16;
  const int half = lane >> 4;
  const int l15 = lane & 15;

  const _Float16* arow = A + (size_t)(m0 + l15) * lda + 8 * half;
  const _Float16* wrow = Wp + (size_t)(n0 + l15) * lda + 8 * half;

  v8f acc = {};
  for (int k0 = 0; k0 < lda; k0 += 32) {
    v8h ca = *(const v8h*)(arow + k0);
    v8h cb = *(const v8h*)(arow + k0 + 16);
    if (FUSED) {
      const int ks = k0 + 8 * half;
      v8h sa = *(const v8h*)(sch + ks);
      v8h sb = *(const v8h*)(sch + ks + 16);
      v8h ha = *(const v8h*)(shh + ks);
      v8h hb = *(const v8h*)(shh + ks + 16);
      v8h z = {};
      ca = __builtin_elementwise_max(ca * sa + ha, z);   // v_pk_fma / v_pk_max
      cb = __builtin_elementwise_max(cb * sb + hb, z);
    }
    v16h a = __builtin_shufflevector(ca, cb, 0, 1, 2, 3, 4, 5, 6, 7,
                                     8, 9, 10, 11, 12, 13, 14, 15);
    v16h b = __builtin_shufflevector(*(const v8h*)(wrow + k0), *(const v8h*)(wrow + k0 + 16),
                                     0, 1, 2, 3, 4, 5, 6, 7, 8, 9, 10, 11, 12, 13, 14, 15);
    acc = __builtin_amdgcn_wmma_f32_16x16x32_f16(false, a, false, b,
                                                 (short)0, acc, false, false);
  }

  const float bsv = bias[n0 + l15];
  #pragma unroll
  for (int rl = 0; rl < 8; ++rl) {
    int rr = m0 + rl + (half << 3);          // D: M = rl + 8*half, N = lane&15
    Y[(size_t)rr * Cout + (n0 + l15)] = (_Float16)(acc[rl] + bsv);
  }
}

// --------------------------- batch-norm stats ------------------------------
__global__ __launch_bounds__(256)
void stats_kernel(const _Float16* __restrict__ Y, long long total, int C,
                  float* __restrict__ gsum, float* __restrict__ gsq) {
  extern __shared__ float sm[];
  float* ssum = sm;
  float* ssq = sm + C;
  for (int c = threadIdx.x; c < 2 * C; c += blockDim.x) sm[c] = 0.0f;
  __syncthreads();
  for (long long i = (long long)blockIdx.x * blockDim.x + threadIdx.x; i < total;
       i += (long long)gridDim.x * blockDim.x) {
    float v = (float)Y[i];
    int c = (int)(i & (long long)(C - 1));
    atomicAdd(&ssum[c], v);
    atomicAdd(&ssq[c], v * v);
  }
  __syncthreads();
  for (int c = threadIdx.x; c < C; c += blockDim.x) {
    atomicAdd(&gsum[c], ssum[c]);
    atomicAdd(&gsq[c], ssq[c]);
  }
}

__global__ void finalize_stats_kernel(const float* __restrict__ gsum,
                                      const float* __restrict__ gsq,
                                      const float* __restrict__ gamma,
                                      const float* __restrict__ beta,
                                      float invR, int C,
                                      float* __restrict__ scale,
                                      float* __restrict__ shift,
                                      _Float16* __restrict__ scaleh,
                                      _Float16* __restrict__ shifth) {
  int c = blockIdx.x * blockDim.x + threadIdx.x;
  if (c >= C) return;
  float mu = gsum[c] * invR;
  float var = gsq[c] * invR - mu * mu;
  float s = gamma[c] * rsqrtf(var + 1e-5f);
  float sh = beta[c] - mu * s;
  scale[c] = s;
  shift[c] = sh;
  scaleh[c] = (_Float16)s;
  shifth[c] = (_Float16)sh;
}

// ------------------------- fused BN+ReLU max-pool --------------------------
__global__ void maxpool_kernel(const _Float16* __restrict__ Y,
                               const float* __restrict__ scale,
                               const float* __restrict__ shift,
                               long long QC, int K, int C,
                               float* __restrict__ out) {
  for (long long i = (long long)blockIdx.x * blockDim.x + threadIdx.x; i < QC;
       i += (long long)gridDim.x * blockDim.x) {
    int c = (int)(i % C);
    long long q = i / C;
    const _Float16* y = Y + (size_t)q * K * C + c;
    float s = scale[c], sh = shift[c];
    float m = 0.0f;   // ReLU outputs are >= 0
    for (int k = 0; k < K; ++k) {
      float v = fmaxf((float)y[(size_t)k * C] * s + sh, 0.0f);
      m = fmaxf(m, v);
    }
    out[i] = m;
  }
}

// ------------------------------ host side ----------------------------------

static inline size_t align256(size_t x) { return (x + 255) & ~(size_t)255; }

struct StatsBufs {
  float* sbuf;      // 2048: sum | sumsq
  float* sscale;    // 1024
  float* sshift;    // 1024
  _Float16* scaleh; // 1024
  _Float16* shifth; // 1024
};

static void run_mlp(hipStream_t stream, const _Float16* G, int CinP, int CinR,
                    const int C[3], long long R,
                    const float* const* Wts, const float* const* Bs,
                    const float* const* Gs, const float* const* Bes,
                    _Float16* const* Wp,
                    _Float16* P0, _Float16* P1, const StatsBufs& sb) {
  const _Float16* A = G;
  int lda = CinP;
  int cin = CinR;
  _Float16* outs[3] = {P0, P1, P0};
  float* gsum = sb.sbuf;
  float* gsq = sb.sbuf + 1024;

  for (int l = 0; l < 3; ++l) {
    int Cout = C[l];
    // weight f32 -> padded f16
    int wn = Cout * lda;
    wprep_kernel<<<(wn + 255) / 256, 256, 0, stream>>>(Wts[l], cin, lda, Cout,
                                                       Wp[l]);
    long long mtiles = R / 16;
    dim3 grid((unsigned)((mtiles + 3) / 4), (unsigned)(Cout / 16));
    if (l == 0)
      gemm_wmma_kernel<false><<<grid, 128, 0, stream>>>(
          A, lda, nullptr, nullptr, Wp[l], Bs[l], (int)R, Cout, outs[l]);
    else
      gemm_wmma_kernel<true><<<grid, 128, 0, stream>>>(
          A, lda, sb.scaleh, sb.shifth, Wp[l], Bs[l], (int)R, Cout, outs[l]);

    zero_f32_kernel<<<(2048 + 255) / 256, 256, 0, stream>>>(sb.sbuf, 2048);
    long long total = R * Cout;
    int blocks = (int)((total + 255) / 256);
    if (blocks > 768) blocks = 768;
    stats_kernel<<<blocks, 256, 2 * Cout * sizeof(float), stream>>>(
        outs[l], total, Cout, gsum, gsq);
    finalize_stats_kernel<<<(Cout + 255) / 256, 256, 0, stream>>>(
        gsum, gsq, Gs[l], Bes[l], 1.0f / (float)R, Cout, sb.sscale, sb.sshift,
        sb.scaleh, sb.shifth);
    A = outs[l];
    lda = cin = Cout;
  }
}

extern "C" void kernel_launch(void* const* d_in, const int* in_sizes, int n_in,
                              void* d_out, int out_size, void* d_ws, size_t ws_size,
                              hipStream_t stream) {
  // ---- input mapping (detect insertion-order vs jax-sorted flattening) ----
  const float* xyz = nullptr;
  const float *w[3][3], *bs[3][3], *gm[3][3], *be[3][3];
  if (in_sizes[0] == BB * 3 * NN) {
    // insertion order: xyz, then per-SA: w[3], b[3], gamma[3], beta[3]
    xyz = (const float*)d_in[0];
    int p = 1;
    for (int s = 0; s < 3; ++s) {
      for (int l = 0; l < 3; ++l) w[s][l]  = (const float*)d_in[p++];
      for (int l = 0; l < 3; ++l) bs[s][l] = (const float*)d_in[p++];
      for (int l = 0; l < 3; ++l) gm[s][l] = (const float*)d_in[p++];
      for (int l = 0; l < 3; ++l) be[s][l] = (const float*)d_in[p++];
    }
  } else {
    // sorted keys: per-SA: b, beta, gamma, w; xyz last
    int p = 0;
    for (int s = 0; s < 3; ++s) {
      for (int l = 0; l < 3; ++l) bs[s][l] = (const float*)d_in[p++];
      for (int l = 0; l < 3; ++l) be[s][l] = (const float*)d_in[p++];
      for (int l = 0; l < 3; ++l) gm[s][l] = (const float*)d_in[p++];
      for (int l = 0; l < 3; ++l) w[s][l]  = (const float*)d_in[p++];
    }
    xyz = (const float*)d_in[p];
  }

  // ---- stage dimensions ----
  const int S1 = 512, K1 = 32,  Cin1 = 6,   Kp1 = 32;
  const int S2 = 128, K2 = 64,  Cin2 = 131, Kp2 = 160;
  const int           K3 = 128, Cin3 = 259, Kp3 = 288;
  const int C1[3] = {64, 64, 128};
  const int C2[3] = {128, 128, 256};
  const int C3[3] = {256, 512, 1024};
  const long long R1 = (long long)BB * S1 * K1;   // 524288
  const long long R2 = (long long)BB * S2 * K2;   // 262144
  const long long R3 = (long long)BB * K3;        // 4096

  // Wp sizes: layer0 uses padded Cin, layers 1/2 use previous Cout.
  const int wpad1[3] = {Kp1, C1[0], C1[1]};
  const int wpad2[3] = {Kp2, C2[0], C2[1]};
  const int wpad3[3] = {Kp3, C3[0], C3[1]};

  // ---- workspace carve ----
  const size_t gmax = (size_t)((R2 * Kp2 > R1 * Kp1) ? R2 * Kp2 : R1 * Kp1);
  const size_t pmax = (size_t)R1 * 128;           // == R2*256, > R3*1024
  char* cur = (char*)d_ws;
  auto carve = [&](size_t bytes) { char* r = cur; cur += align256(bytes); return (void*)r; };

  float*    pts    = (float*)carve((size_t)BB * NN * 3 * 4);
  int*      fps1   = (int*)carve((size_t)BB * S1 * 4);
  float*    l1xyz  = (float*)carve((size_t)BB * S1 * 3 * 4);
  int*      ball1  = (int*)carve((size_t)BB * S1 * K1 * 4);
  float*    l1pts  = (float*)carve((size_t)BB * S1 * 128 * 4);
  int*      fps2   = (int*)carve((size_t)BB * S2 * 4);
  float*    l2xyz  = (float*)carve((size_t)BB * S2 * 3 * 4);
  int*      ball2  = (int*)carve((size_t)BB * S2 * K2 * 4);
  float*    l2pts  = (float*)carve((size_t)BB * S2 * 256 * 4);
  _Float16* G      = (_Float16*)carve(gmax * 2);
  _Float16* P0     = (_Float16*)carve(pmax * 2);
  _Float16* P1     = (_Float16*)carve(pmax * 2);

  _Float16* Wp1[3]; _Float16* Wp2[3]; _Float16* Wp3[3];
  for (int l = 0; l < 3; ++l) Wp1[l] = (_Float16*)carve((size_t)C1[l] * wpad1[l] * 2);
  for (int l = 0; l < 3; ++l) Wp2[l] = (_Float16*)carve((size_t)C2[l] * wpad2[l] * 2);
  for (int l = 0; l < 3; ++l) Wp3[l] = (_Float16*)carve((size_t)C3[l] * wpad3[l] * 2);

  StatsBufs sb;
  sb.sbuf   = (float*)carve(2048 * 4);
  sb.sscale = (float*)carve(1024 * 4);
  sb.sshift = (float*)carve(1024 * 4);
  sb.scaleh = (_Float16*)carve(1024 * 2);
  sb.shifth = (_Float16*)carve(1024 * 2);
  if ((size_t)(cur - (char*)d_ws) > ws_size) return;  // workspace too small

  // ---- SA1 ----
  transpose_xyz_kernel<<<(BB * 3 * NN + 255) / 256, 256, 0, stream>>>(xyz, pts);
  fps_kernel<<<BB, 256, 0, stream>>>(pts, NN, S1, fps1, l1xyz);
  ball_query_kernel<<<BB * S1, 32, 0, stream>>>(pts, l1xyz, NN, S1, K1,
                                                0.2f * 0.2f, ball1);
  group_kernel<<<2048, 256, 0, stream>>>(pts, pts, 3, l1xyz, ball1,
                                         S1, K1, Kp1, NN, R1, G);
  run_mlp(stream, G, Kp1, Cin1, C1, R1, w[0], bs[0], gm[0], be[0], Wp1,
          P0, P1, sb);
  {
    long long QC = (long long)BB * S1 * 128;
    maxpool_kernel<<<(unsigned)((QC + 255) / 256), 256, 0, stream>>>(
        P0, sb.sscale, sb.sshift, QC, K1, 128, l1pts);
  }

  // ---- SA2 ----
  fps_kernel<<<BB, 256, 0, stream>>>(l1xyz, S1, S2, fps2, l2xyz);
  ball_query_kernel<<<BB * S2, 32, 0, stream>>>(l1xyz, l2xyz, S1, S2, K2,
                                                0.4f * 0.4f, ball2);
  group_kernel<<<2048, 256, 0, stream>>>(l1xyz, l1pts, 128, l2xyz, ball2,
                                         S2, K2, Kp2, S1, R2, G);
  run_mlp(stream, G, Kp2, Cin2, C2, R2, w[1], bs[1], gm[1], be[1], Wp2,
          P0, P1, sb);
  {
    long long QC = (long long)BB * S2 * 256;
    maxpool_kernel<<<(unsigned)((QC + 255) / 256), 256, 0, stream>>>(
        P0, sb.sscale, sb.sshift, QC, K2, 256, l2pts);
  }

  // ---- SA3 (group_all) ----
  group_all_kernel<<<64, 256, 0, stream>>>(l2xyz, l2pts, 256, Kp3, R3, G);
  run_mlp(stream, G, Kp3, Cin3, C3, R3, w[2], bs[2], gm[2], be[2], Wp3,
          P0, P1, sb);
  {
    long long QC = (long long)BB * 1024;   // == out_size
    maxpool_kernel<<<(unsigned)((QC + 255) / 256), 256, 0, stream>>>(
        P0, sb.sscale, sb.sshift, QC, K3, 1024, (float*)d_out);
  }
}
